// CSA_model_23639499997806
// MI455X (gfx1250) — matrix operand order repaired and checked by reading, not compiled, not run on hardware
//
#include <hip/hip_runtime.h>
#include <stdint.h>

#define EPSF 1e-8f
constexpr int Bn = 4;
constexpr int Cc = 512;
constexpr int Nn = 4096;   // 64*64

typedef __attribute__((ext_vector_type(16))) __bf16 v16bf;
typedef __attribute__((ext_vector_type(8)))  float  v8f;

union BfOp { v16bf v; uint32_t u[8]; };

__device__ __forceinline__ uint16_t f2bf(float f) {
    uint32_t u = __float_as_uint(f);
    uint32_t r = (u + 0x7FFFu + ((u >> 16) & 1u)) >> 16;   // RNE
    return (uint16_t)r;
}

// ---------------------------------------------------------------------------
// Kernel 1: per-pixel feature norms.  norms[b*N+n] = ||x[b,:,n]||2
// ---------------------------------------------------------------------------
__global__ __launch_bounds__(256) void k_norms(const float* __restrict__ x,
                                               float* __restrict__ norms) {
    int t = blockIdx.x * 256 + threadIdx.x;        // over B*N
    int b = t >> 12;                               // N = 4096
    int n = t & (Nn - 1);
    const float* xb = x + (size_t)b * Cc * Nn + n;
    float acc = 0.f;
    #pragma unroll 8
    for (int c = 0; c < Cc; ++c) {
        float v = xb[(size_t)c * Nn];
        acc += v * v;
    }
    norms[t] = sqrtf(acc);
}

// ---------------------------------------------------------------------------
// Kernel 2: fnA[b,n,c] = bf16( x[b,c,n] / (norm+eps) )  (LDS-tiled transpose)
// ---------------------------------------------------------------------------
__global__ __launch_bounds__(256) void k_fnA(const float* __restrict__ x,
                                             const float* __restrict__ norms,
                                             uint16_t* __restrict__ fnA) {
    __shared__ float tile[32][33];
    int n0 = blockIdx.x * 32;
    int c0 = blockIdx.y * 32;
    int b  = blockIdx.z;
    int tx = threadIdx.x;          // 0..31
    int ty = threadIdx.y;          // 0..7
    float rn = 1.f / (norms[(size_t)b * Nn + n0 + tx] + EPSF);
    #pragma unroll
    for (int cy = ty; cy < 32; cy += 8)
        tile[cy][tx] = x[((size_t)b * Cc + c0 + cy) * Nn + n0 + tx] * rn;
    __syncthreads();
    #pragma unroll
    for (int ny = ty; ny < 32; ny += 8)
        fnA[((size_t)b * Nn + n0 + ny) * Cc + c0 + tx] = f2bf(tile[tx][ny]);
}

// ---------------------------------------------------------------------------
// Kernel 3: fused sim-GEMM + masked row max/argmax.
// Block = 4 waves; block handles (b, 16-row tile); waves split 256 col tiles.
// ---------------------------------------------------------------------------
__global__ __launch_bounds__(128) void k_argmax(const uint16_t* __restrict__ fnA,
                                                const int* __restrict__ mask,
                                                float* __restrict__ dmax,
                                                int* __restrict__ aidx) {
    __shared__ float sV[4][16];
    __shared__ int   sI[4][16];

    int blk   = blockIdx.x;            // b*256 + ntile
    int b     = blk >> 8;
    int n0    = (blk & 255) * 16;
    int lane  = threadIdx.x & 31;
    int wave  = threadIdx.x >> 5;
    int halfH = lane >> 4;             // 0: lanes 0-15, 1: lanes 16-31
    int l15   = lane & 15;

    const uint16_t* base = fnA + (size_t)b * Nn * Cc;

    // Resident A strip: rows n0..n0+15, all K=512, as 16 K-chunks of 32.
    // 16-bit A 16x32 layout: lane<16 holds K {kc*32+0..7, kc*32+16..23},
    // lane>=16 holds K {kc*32+8..15, kc*32+24..31} of row (lane&15).
    BfOp a[16];
    {
        const uint16_t* arow = base + (size_t)(n0 + l15) * Cc;
        #pragma unroll
        for (int kc = 0; kc < 16; ++kc) {
            int kb = kc * 32 + halfH * 8;
            uint4 lo = *(const uint4*)(arow + kb);
            uint4 hi = *(const uint4*)(arow + kb + 16);
            a[kc].u[0] = lo.x; a[kc].u[1] = lo.y; a[kc].u[2] = lo.z; a[kc].u[3] = lo.w;
            a[kc].u[4] = hi.x; a[kc].u[5] = hi.y; a[kc].u[6] = hi.z; a[kc].u[7] = hi.w;
        }
    }

    float bestV[8];
    int   bestI[8];
    #pragma unroll
    for (int i = 0; i < 8; ++i) { bestV[i] = -3.0e38f; bestI[i] = 0x7fffffff; }

    for (int mt = wave; mt < 256; mt += 4) {
        int mcol = mt * 16 + l15;      // this lane's sim column for all 8 accs
        // B operand (32x16 bf16): lanes 0-15 hold K {0..7,16..23}, lanes 16-31 K {8..15,24..31}
        // of column m = lane&15 -> two 8-wide c runs of row mcol in fnA.
        const uint16_t* brow = base + (size_t)mcol * Cc;
        v8f acc = {};
        #pragma unroll
        for (int kc = 0; kc < 16; ++kc) {
            BfOp bo;
            int kb = kc * 32 + halfH * 8;
            uint4 lo = *(const uint4*)(brow + kb);
            uint4 hi = *(const uint4*)(brow + kb + 16);
            bo.u[0] = lo.x; bo.u[1] = lo.y; bo.u[2] = lo.z; bo.u[3] = lo.w;
            bo.u[4] = hi.x; bo.u[5] = hi.y; bo.u[6] = hi.z; bo.u[7] = hi.w;
            acc = __builtin_amdgcn_wmma_f32_16x16x32_bf16(
                false, a[kc].v, false, bo.v, (short)0, acc, false, false);
        }
        bool hole = (mask[mcol] != 0);
        #pragma unroll
        for (int i = 0; i < 8; ++i) {
            float v = hole ? -1e30f : acc[i];
            if (v > bestV[i] || (v == bestV[i] && mcol < bestI[i])) {
                bestV[i] = v; bestI[i] = mcol;
            }
        }
    }

    // reduce over the 16 lanes of each half (same row per half):
    // D layout: reg i, lanes 0-15 -> row i, lanes 16-31 -> row i+8; col = lane&15
    #pragma unroll
    for (int i = 0; i < 8; ++i) {
        float v = bestV[i]; int id = bestI[i];
        #pragma unroll
        for (int m = 1; m < 16; m <<= 1) {
            float ov = __shfl_xor(v, m, 32);
            int   oi = __shfl_xor(id, m, 32);
            if (ov > v || (ov == v && oi < id)) { v = ov; id = oi; }
        }
        bestV[i] = v; bestI[i] = id;
    }
    if (l15 == 0) {                    // lanes 0 and 16 publish rows 0-7 / 8-15
        #pragma unroll
        for (int i = 0; i < 8; ++i) {
            sV[wave][i + 8 * halfH] = bestV[i];
            sI[wave][i + 8 * halfH] = bestI[i];
        }
    }
    __syncthreads();
    if (threadIdx.x < 16) {
        int r = threadIdx.x;
        float v = sV[0][r]; int id = sI[0][r];
        #pragma unroll
        for (int w = 1; w < 4; ++w) {
            float ov = sV[w][r]; int oi = sI[w][r];
            if (ov > v || (ov == v && oi < id)) { v = ov; id = oi; }
        }
        dmax[(size_t)b * Nn + n0 + r] = fmaxf(v, 0.f);
        aidx[(size_t)b * Nn + n0 + r] = id;
    }
}

// ---------------------------------------------------------------------------
// Kernel 4: sequential coherent pass. One wave per batch; lane owns 16 chans.
// ---------------------------------------------------------------------------
__global__ __launch_bounds__(32) void k_scan(const float* __restrict__ x,
                                             const int* __restrict__ mask,
                                             const float* __restrict__ norms,
                                             const float* __restrict__ dmax,
                                             const int* __restrict__ aidx,
                                             float* __restrict__ out) {
    int b = blockIdx.x;
    int lane = threadIdx.x;
    const float* xb = x + (size_t)b * Cc * Nn;
    float* ob = out + (size_t)b * Cc * Nn;

    float prev[16];
    #pragma unroll
    for (int j = 0; j < 16; ++j) prev[j] = 0.f;

    for (int n = 0; n < Nn; ++n) {
        if ((n & 15) == 0 && n + 16 < Nn) {      // hint upcoming cache lines
            #pragma unroll
            for (int j = 0; j < 16; ++j)
                __builtin_prefetch(&xb[(size_t)(lane + 32 * j) * Nn + n + 16], 0, 3);
        }
        float feat[16];
        #pragma unroll
        for (int j = 0; j < 16; ++j)
            feat[j] = xb[(size_t)(lane + 32 * j) * Nn + n];

        float gen[16];
        if (mask[n] != 0) {                      // hole: blended generation
            float s1 = 0.f, s2 = 0.f;
            #pragma unroll
            for (int j = 0; j < 16; ++j) {
                s1 += prev[j] * prev[j];
                s2 += prev[j] * feat[j];
            }
            #pragma unroll
            for (int m = 1; m < 32; m <<= 1) {
                s1 += __shfl_xor(s1, m, 32);
                s2 += __shfl_xor(s2, m, 32);
            }
            float rn    = 1.f / (norms[(size_t)b * Nn + n] + EPSF);
            float pnorm = sqrtf(s1);
            float dad   = fmaxf(s2 * rn / (pnorm + EPSF), 0.f);
            float dm    = dmax[(size_t)b * Nn + n];
            int   mi    = aidx[(size_t)b * Nn + n];
            float inv   = 1.f / (dm + dad + EPSF);
            #pragma unroll
            for (int j = 0; j < 16; ++j) {
                float mt = xb[(size_t)(lane + 32 * j) * Nn + mi];
                gen[j] = (dm * mt + dad * prev[j]) * inv;
            }
        } else {                                 // known pixel passes through
            #pragma unroll
            for (int j = 0; j < 16; ++j) gen[j] = feat[j];
        }
        #pragma unroll
        for (int j = 0; j < 16; ++j) {
            ob[(size_t)(lane + 32 * j) * Nn + n] = gen[j];
            prev[j] = gen[j];
        }
    }
}

// ---------------------------------------------------------------------------
extern "C" void kernel_launch(void* const* d_in, const int* in_sizes, int n_in,
                              void* d_out, int out_size, void* d_ws, size_t ws_size,
                              hipStream_t stream) {
    const float* x    = (const float*)d_in[0];   // [4,512,64,64] f32
    const int*   mask = (const int*)d_in[1];     // [64,64] i32
    float* out = (float*)d_out;                  // [4,512,64,64] f32

    char* ws = (char*)d_ws;
    float*     norms = (float*)ws;                         //  64 KB
    float*     dmaxp = (float*)(ws + 1 * 65536);           //  64 KB
    int*       aidx  = (int*)  (ws + 2 * 65536);           //  64 KB
    uint16_t*  fnA   = (uint16_t*)(ws + 3 * 65536);        //  16.8 MB [B,N,C] bf16

    k_norms <<<(Bn * Nn) / 256, 256, 0, stream>>>(x, norms);
    k_fnA   <<<dim3(Nn / 32, Cc / 32, Bn), dim3(32, 8), 0, stream>>>(x, norms, fnA);
    k_argmax<<<Bn * (Nn / 16), 128, 0, stream>>>(fnA, mask, dmaxp, aidx);
    k_scan  <<<Bn, 32, 0, stream>>>(x, mask, norms, dmaxp, aidx, out);
}
